// BotRGCN_Original_32495722562032
// MI455X (gfx1250) — matrix operand order repaired and checked by reading, not compiled, not run on hardware
//
#include <hip/hip_runtime.h>

typedef __attribute__((ext_vector_type(16))) _Float16 v16h;
typedef __attribute__((ext_vector_type(8)))  float    v8f;

__device__ __forceinline__ float lrelu(float x) { return x > 0.f ? x : 0.01f * x; }

// A-fragment K index for 16-bit A 16x32 layout (ISA 7.12.2):
// lanes 0-15 (half=0): elems 0-7 -> K 0-7, elems 8-15 -> K 16-23
// lanes 16-31 (half=1): elems 0-7 -> K 8-15, elems 8-15 -> K 24-31
__device__ __forceinline__ int a_k(int e, int half) {
    return (e & 7) + ((e >> 3) << 4) + (half << 3);
}

__device__ __forceinline__ v8f wmma_f16(v16h a, v16h b, v8f c) {
    return __builtin_amdgcn_wmma_f32_16x16x32_f16(false, a, false, b, (short)0, c, false, false);
}

__device__ __forceinline__ void pack8(v16h& v, int off, float4 x, float4 y) {
    v[off + 0] = (_Float16)x.x; v[off + 1] = (_Float16)x.y;
    v[off + 2] = (_Float16)x.z; v[off + 3] = (_Float16)x.w;
    v[off + 4] = (_Float16)y.x; v[off + 5] = (_Float16)y.y;
    v[off + 6] = (_Float16)y.z; v[off + 7] = (_Float16)y.w;
}

// Load B fragment (32x16 tile, row stride 32): N = lane%16, elems = K in a_k order.
__device__ __forceinline__ v16h load_b32x16(const float* __restrict__ W, int lane) {
    const int bn = lane & 15, bh = lane >> 4;
    v16h b;
#pragma unroll
    for (int e = 0; e < 16; ++e)
        b[e] = (_Float16)W[a_k(e, bh) * 32 + bn];
    return b;
}

// Native no-return fp32 atomic add (device scope, relaxed): global_atomic_add_f32
__device__ __forceinline__ void atomic_add_f32(float* p, float v) {
    (void)__hip_atomic_fetch_add(p, v, __ATOMIC_RELAXED, __HIP_MEMORY_SCOPE_AGENT);
}

// ---------------------------------------------------------------------------
// Kernel 1: fused modality projections + concat + input linear (all WMMA)
// ---------------------------------------------------------------------------
__global__ __launch_bounds__(256)
void embed_kernel(const float* __restrict__ des, const float* __restrict__ tweet,
                  const float* __restrict__ nump, const float* __restrict__ catp,
                  const float* __restrict__ Wd, const float* __restrict__ bd,
                  const float* __restrict__ Wt, const float* __restrict__ bt,
                  const float* __restrict__ Wn, const float* __restrict__ bn,
                  const float* __restrict__ Wc, const float* __restrict__ bc,
                  const float* __restrict__ Wi, const float* __restrict__ bi,
                  float* __restrict__ x0, int N)
{
    __shared__ float lds[8][16][32];
    const int lane = threadIdx.x & 31;
    const int wave = threadIdx.x >> 5;
    const int tile = blockIdx.x * 8 + wave;
    const int tiles = (N + 15) >> 4;
    if (tile >= tiles) return;                  // wave-uniform
    const int m = lane & 15, half = lane >> 4;
    int row = tile * 16 + m;
    if (row >= N) row = N - 1;
    const int n = lane & 15;

    v8f cd = {}; v8f ct = {};
    for (int k0 = 0; k0 < 768; k0 += 32) {
        // A from des
        v16h ad;
        {
            const float4* p = reinterpret_cast<const float4*>(des + (size_t)row * 768 + k0 + half * 8);
            pack8(ad, 0, p[0], p[1]);           // K = k0 + half*8 + 0..7
            pack8(ad, 8, p[4], p[5]);           // K = k0 + 16 + half*8 + 0..7
        }
        // B from Wd (768x8, pad N to 16 with zeros)
        v16h bd_;
        {
#pragma unroll
            for (int e = 0; e < 16; ++e) {
                float w = Wd[(size_t)(k0 + a_k(e, half)) * 8 + (n & 7)];
                bd_[e] = (_Float16)(n < 8 ? w : 0.f);
            }
        }
        cd = wmma_f16(ad, bd_, cd);

        v16h at;
        {
            const float4* p = reinterpret_cast<const float4*>(tweet + (size_t)row * 768 + k0 + half * 8);
            pack8(at, 0, p[0], p[1]);
            pack8(at, 8, p[4], p[5]);
        }
        v16h bt_;
        {
#pragma unroll
            for (int e = 0; e < 16; ++e) {
                float w = Wt[(size_t)(k0 + a_k(e, half)) * 8 + (n & 7)];
                bt_[e] = (_Float16)(n < 8 ? w : 0.f);
            }
        }
        ct = wmma_f16(at, bt_, ct);
    }

    // num_prop (K=6) and cat_prop (K=3): single zero-padded WMMA each.
    // Unconditional clamped-index loads + selects (no exec-mask branches).
    const int lc6 = (lane < 6 ? lane : 5);
    const int lc3 = (lane < 3 ? lane : 2);
    v16h an, bn_, ac, bc_;
#pragma unroll
    for (int e = 0; e < 16; ++e) {
        int k = a_k(e, half);
        float vn = nump[(size_t)row * 6 + (k < 6 ? k : 5)];
        float vc = catp[(size_t)row * 3 + (k < 3 ? k : 2)];
        an[e] = (_Float16)(k < 6 ? vn : 0.f);
        ac[e] = (_Float16)(k < 3 ? vc : 0.f);
        float wn = Wn[(k < 6 ? k : 5) * 8 + (n & 7)];
        float wc = Wc[(k < 3 ? k : 2) * 8 + (n & 7)];
        bn_[e] = (_Float16)((k < 6 && n < 8) ? wn : 0.f);
        bc_[e] = (_Float16)((k < 3 && n < 8) ? wc : 0.f);
    }
    v8f z = {};
    v8f cn = wmma_f16(an, bn_, z);
    v8f cc = wmma_f16(ac, bc_, z);

    // lrelu + concat into per-wave LDS tile [16 nodes][32 feats]
    if (n < 8) {
#pragma unroll
        for (int r = 0; r < 8; ++r) {
            int mm = r + half * 8;
            lds[wave][mm][n]      = lrelu(cd[r] + bd[n]);
            lds[wave][mm][8 + n]  = lrelu(ct[r] + bt[n]);
            lds[wave][mm][16 + n] = lrelu(cn[r] + bn[n]);
            lds[wave][mm][24 + n] = lrelu(cc[r] + bc[n]);
        }
    }
    // intra-wave LDS store->load: DS ops from same wave are in-order

    // x = lrelu(x_pre @ Wi + bi): A from LDS, 2 N-tiles of Wi
    v16h ax;
#pragma unroll
    for (int e = 0; e < 16; ++e)
        ax[e] = (_Float16)lds[wave][m][a_k(e, half)];
#pragma unroll
    for (int nt = 0; nt < 2; ++nt) {
        v16h bw = load_b32x16(Wi + nt * 16, lane);
        v8f zz = {};
        v8f c = wmma_f16(ax, bw, zz);
        int feat = nt * 16 + n;
#pragma unroll
        for (int r = 0; r < 8; ++r) {
            int node = tile * 16 + r + half * 8;
            if (node < N)
                x0[(size_t)node * 32 + feat] = lrelu(c[r] + bi[feat]);
        }
    }
}

// ---------------------------------------------------------------------------
// Kernel 2: zero fill
// ---------------------------------------------------------------------------
__global__ void zero_kernel(float* __restrict__ p, long long n) {
    long long i = (long long)blockIdx.x * blockDim.x + threadIdx.x;
    if (i < n) p[i] = 0.f;
}

// ---------------------------------------------------------------------------
// Kernel 3: per-relation edge scatter (one wave per edge, one feat per lane)
// ---------------------------------------------------------------------------
__global__ __launch_bounds__(256)
void scatter_kernel(const float* __restrict__ x, const int* __restrict__ src,
                    const int* __restrict__ dst, const int* __restrict__ etype,
                    float* __restrict__ sum, float* __restrict__ cnt, int N, int E)
{
    int e = blockIdx.x * 8 + (threadIdx.x >> 5);
    if (e >= E) return;                         // wave-uniform
    int lane = threadIdx.x & 31;
    int s = src[e], d = dst[e], r = etype[e];
    float v = x[(size_t)s * 32 + lane];
    atomic_add_f32(&sum[((size_t)r * N + d) * 32 + lane], v);
    if (lane == 0) atomic_add_f32(&cnt[(size_t)r * N + d], 1.0f);
}

// ---------------------------------------------------------------------------
// Kernel 4: RGCN combine: out = x@Wroot + mean_r@Wrel[r] + brgcn (3 chained WMMA)
// ---------------------------------------------------------------------------
__global__ __launch_bounds__(256)
void combine_kernel(const float* __restrict__ x, const float* __restrict__ sum,
                    const float* __restrict__ cnt, const float* __restrict__ Wroot,
                    const float* __restrict__ Wrel, const float* __restrict__ brgcn,
                    float* __restrict__ out, int N)
{
    const int lane = threadIdx.x & 31;
    const int wave = threadIdx.x >> 5;
    const int tile = blockIdx.x * 8 + wave;
    const int tiles = (N + 15) >> 4;
    if (tile >= tiles) return;
    const int m = lane & 15, half = lane >> 4, n = lane & 15;
    int row = tile * 16 + m;
    if (row >= N) row = N - 1;

    float inv0 = 1.f / fmaxf(cnt[row], 1.f);
    float inv1 = 1.f / fmaxf(cnt[(size_t)N + row], 1.f);

    v16h ax, a0, a1;
#pragma unroll
    for (int e = 0; e < 16; ++e) {
        int k = a_k(e, half);
        ax[e] = (_Float16)x[(size_t)row * 32 + k];
        a0[e] = (_Float16)(sum[(size_t)row * 32 + k] * inv0);
        a1[e] = (_Float16)(sum[((size_t)N + row) * 32 + k] * inv1);
    }
#pragma unroll
    for (int nt = 0; nt < 2; ++nt) {
        v16h br = load_b32x16(Wroot + nt * 16, lane);
        v16h b0 = load_b32x16(Wrel + nt * 16, lane);
        v16h b1 = load_b32x16(Wrel + 1024 + nt * 16, lane);
        v8f c = {};
        c = wmma_f16(ax, br, c);
        c = wmma_f16(a0, b0, c);
        c = wmma_f16(a1, b1, c);
        int feat = nt * 16 + n;
#pragma unroll
        for (int r = 0; r < 8; ++r) {
            int node = tile * 16 + r + half * 8;
            if (node < N)
                out[(size_t)node * 32 + feat] = c[r] + brgcn[feat];
        }
    }
}

// ---------------------------------------------------------------------------
// Kernel 5: fused head: h = lrelu(x @ Wo1 + bo1); out = h @ Wo2 + bo2
// (2 WMMAs for Wo1, LDS transpose, 1 zero-padded WMMA for Wo2)
// ---------------------------------------------------------------------------
__global__ __launch_bounds__(256)
void head_kernel(const float* __restrict__ x, const float* __restrict__ Wo1,
                 const float* __restrict__ bo1, const float* __restrict__ Wo2,
                 const float* __restrict__ bo2, float* __restrict__ out, int N)
{
    __shared__ float lds[8][16][32];
    const int lane = threadIdx.x & 31;
    const int wave = threadIdx.x >> 5;
    const int tile = blockIdx.x * 8 + wave;
    const int tiles = (N + 15) >> 4;
    if (tile >= tiles) return;
    const int m = lane & 15, half = lane >> 4, n = lane & 15;
    int row = tile * 16 + m;
    if (row >= N) row = N - 1;

    v16h ax;
#pragma unroll
    for (int e = 0; e < 16; ++e)
        ax[e] = (_Float16)x[(size_t)row * 32 + a_k(e, half)];
#pragma unroll
    for (int nt = 0; nt < 2; ++nt) {
        v16h bw = load_b32x16(Wo1 + nt * 16, lane);
        v8f z = {};
        v8f c = wmma_f16(ax, bw, z);
        int feat = nt * 16 + n;
#pragma unroll
        for (int r = 0; r < 8; ++r)
            lds[wave][r + half * 8][feat] = lrelu(c[r] + bo1[feat]);
    }

    // final 32 -> 2 linear as one WMMA with Wo2 zero-padded to 32x16
    v16h ah;
#pragma unroll
    for (int e = 0; e < 16; ++e)
        ah[e] = (_Float16)lds[wave][m][a_k(e, half)];
    v16h b2;
#pragma unroll
    for (int e = 0; e < 16; ++e) {
        float w = Wo2[a_k(e, half) * 2 + (n & 1)];
        b2[e] = (_Float16)(n < 2 ? w : 0.f);
    }
    v8f z2 = {};
    v8f c2 = wmma_f16(ah, b2, z2);
    if (n < 2) {
#pragma unroll
        for (int r = 0; r < 8; ++r) {
            int node = tile * 16 + r + half * 8;
            if (node < N)
                out[(size_t)node * 2 + n] = c2[r] + bo2[n];
        }
    }
}

// ---------------------------------------------------------------------------
extern "C" void kernel_launch(void* const* d_in, const int* in_sizes, int n_in,
                              void* d_out, int out_size, void* d_ws, size_t ws_size,
                              hipStream_t stream)
{
    const float* des   = (const float*)d_in[0];
    const float* tweet = (const float*)d_in[1];
    const float* nump  = (const float*)d_in[2];
    const float* catp  = (const float*)d_in[3];
    const int*   eidx  = (const int*)d_in[4];
    const int*   etype = (const int*)d_in[5];
    const float* Wd = (const float*)d_in[6],  *bd = (const float*)d_in[7];
    const float* Wt = (const float*)d_in[8],  *bt = (const float*)d_in[9];
    const float* Wn = (const float*)d_in[10], *bn = (const float*)d_in[11];
    const float* Wc = (const float*)d_in[12], *bc = (const float*)d_in[13];
    const float* Wi = (const float*)d_in[14], *bi = (const float*)d_in[15];
    const float* Wrel  = (const float*)d_in[16];
    const float* Wroot = (const float*)d_in[17];
    const float* brgcn = (const float*)d_in[18];
    const float* Wo1 = (const float*)d_in[19], *bo1 = (const float*)d_in[20];
    const float* Wo2 = (const float*)d_in[21], *bo2 = (const float*)d_in[22];
    float* out = (float*)d_out;

    const int N = in_sizes[2] / 6;
    const int E = in_sizes[5];
    const int* src = eidx;
    const int* dst = eidx + E;

    float* x0  = (float*)d_ws;                        // [N,32]
    float* x1  = x0 + (size_t)N * 32;                 // [N,32]
    float* sum = x1 + (size_t)N * 32;                 // [2,N,32]
    float* cnt = sum + (size_t)2 * N * 32;            // [2,N]
    const long long zn = (long long)2 * N * 32 + (long long)2 * N;

    const int tiles = (N + 15) / 16;
    const int tb = (tiles + 7) / 8;
    const int zb = (int)((zn + 255) / 256);
    const int eb = (E + 7) / 8;

    // embed + concat + input linear
    embed_kernel<<<tb, 256, 0, stream>>>(des, tweet, nump, catp,
                                         Wd, bd, Wt, bt, Wn, bn, Wc, bc,
                                         Wi, bi, x0, N);

    // RGCN conv 1: x0 -> x1
    zero_kernel<<<zb, 256, 0, stream>>>(sum, zn);
    scatter_kernel<<<eb, 256, 0, stream>>>(x0, src, dst, etype, sum, cnt, N, E);
    combine_kernel<<<tb, 256, 0, stream>>>(x0, sum, cnt, Wroot, Wrel, brgcn, x1, N);

    // RGCN conv 2: x1 -> x0
    zero_kernel<<<zb, 256, 0, stream>>>(sum, zn);
    scatter_kernel<<<eb, 256, 0, stream>>>(x1, src, dst, etype, sum, cnt, N, E);
    combine_kernel<<<tb, 256, 0, stream>>>(x1, sum, cnt, Wroot, Wrel, brgcn, x0, N);

    // fused output head
    head_kernel<<<tb, 256, 0, stream>>>(x0, Wo1, bo1, Wo2, bo2, out, N);
}
